// PLTBlock_88364657148296
// MI455X (gfx1250) — matrix operand order repaired and checked by reading, not compiled, not run on hardware
//
#include <hip/hip_runtime.h>

typedef __attribute__((ext_vector_type(16))) __bf16 v16bf;
typedef __attribute__((ext_vector_type(8)))  float  v8f;
typedef __attribute__((ext_vector_type(4)))  int    v4i_t;

__device__ __forceinline__ unsigned short f2bf(float f) {
  return __builtin_bit_cast(unsigned short, (__bf16)f);
}

union FragBF { uint4 q[2]; v16bf v; };

#if defined(__has_builtin)
#if __has_builtin(__builtin_amdgcn_global_load_async_to_lds_b128)
#define ASYNC_LDS_COPY 1
#endif
#endif

typedef __attribute__((address_space(1))) v4i_t gas_v4i;  // global int4
typedef __attribute__((address_space(3))) v4i_t las_v4i;  // LDS int4

__device__ __forceinline__ void async_cp16(const unsigned short* g, unsigned short* l) {
#ifdef ASYNC_LDS_COPY
  __builtin_amdgcn_global_load_async_to_lds_b128(
      (gas_v4i*)(unsigned long long)g,
      (las_v4i*)(unsigned)(unsigned long long)l, 0, 0);
#else
  (void)g; (void)l;
#endif
}

__device__ __forceinline__ void wait_async0() {
#if defined(__has_builtin)
#if __has_builtin(__builtin_amdgcn_s_wait_asynccnt)
  __builtin_amdgcn_s_wait_asynccnt(0);
#else
  asm volatile("s_wait_asynccnt 0x0" ::: "memory");
#endif
#else
  asm volatile("s_wait_asynccnt 0x0" ::: "memory");
#endif
}

// ---------------------------------------------------------------------------
// fp32 -> bf16 conversion, 4 elements/thread/iter
// ---------------------------------------------------------------------------
__global__ void cvt_f32_to_bf16_x4(const float4* __restrict__ in,
                                   uint2* __restrict__ out, long long n4) {
  long long i      = (long long)blockIdx.x * blockDim.x + threadIdx.x;
  long long stride = (long long)gridDim.x * blockDim.x;
  for (; i < n4; i += stride) {
    float4 f = in[i];
    uint2 o;
    o.x = (unsigned)f2bf(f.x) | ((unsigned)f2bf(f.y) << 16);
    o.y = (unsigned)f2bf(f.z) | ((unsigned)f2bf(f.w) << 16);
    out[i] = o;
  }
}

// ---------------------------------------------------------------------------
// C = A (MxK, row-major bf16) * Bm^T  (Bm is NxK row-major bf16), fp32 accum.
// Block 128(M) x 256(N), K-step 32, 8 waves (2x4), wave tile 64x64 (4x4 WMMA).
// Double-buffered LDS (rows padded to 80B for conflict-free b128 reads);
// async global->LDS copies when available.
// EPI 0: bf16 row-major   out[m*ldc + n]
// EPI 1: bf16 transposed  out[n*ldc + m]   (contiguous 16B store per tile)
// EPI 2: fp32 NT          out[m*ldc+n] = c + bp[n] + Xres[m*ldc+n]
// M = gridDim.y*128, N = gridDim.x*256, K multiple of 32.
// ---------------------------------------------------------------------------
#define BM 128
#define BN 256
#define BK 32
#define BKP 40                // padded LDS row stride in elements (80 B: 20 banks)
#define LDS_A (BM * BKP)      // elements per buffer
#define LDS_B (BN * BKP)

template <int EPI>
__global__ __launch_bounds__(256)
void gemm_bf16_wmma(const unsigned short* __restrict__ A,
                    const unsigned short* __restrict__ Bm,
                    void* __restrict__ OutP,
                    const float* __restrict__ Xres,
                    const float* __restrict__ bp,
                    int K, int lda, int ldb, int ldc,
                    long long sA, long long sB, long long sC) {
  __shared__ __align__(16) unsigned short lA[2 * LDS_A];
  __shared__ __align__(16) unsigned short lB[2 * LDS_B];

  const int tid  = threadIdx.x;
  const int lane = tid & 31;
  const int w    = tid >> 5;   // 0..7
  const int wm   = w & 1;      // 64-row half of the block
  const int wn   = w >> 1;     // 0..3 -> 64-col quarter
  const int lhi  = lane >> 4;  // 0/1 half-wave
  const int llo  = lane & 15;

  const long long bz = blockIdx.z;
  const unsigned short* Ab = A + bz * sA + (long long)(blockIdx.y * BM) * lda;
  const unsigned short* Bb = Bm + bz * sB + (long long)(blockIdx.x * BN) * ldb;

  v8f acc[4][4];
#pragma unroll
  for (int mi = 0; mi < 4; ++mi)
#pragma unroll
    for (int ni = 0; ni < 4; ++ni)
      acc[mi][ni] = (v8f){0.f, 0.f, 0.f, 0.f, 0.f, 0.f, 0.f, 0.f};

  const int bRow0 = tid;  // B: one full 32-elem row per thread (4 x 16B)
  const int nsteps = K >> 5;

#ifdef ASYNC_LDS_COPY
  // ---- async double-buffered pipeline ----
#pragma unroll
  for (int j = 0; j < 2; ++j) {
    const int row = ((tid * 2 + j) >> 2), seg = ((tid * 2 + j) & 3);
    async_cp16(Ab + (long long)row * lda + seg * 8, lA + row * BKP + seg * 8);
  }
#pragma unroll
  for (int j = 0; j < 4; ++j)
    async_cp16(Bb + (long long)bRow0 * ldb + j * 8, lB + bRow0 * BKP + j * 8);
  wait_async0();
  __syncthreads();

  for (int step = 0; step < nsteps; ++step) {
    const int cur = step & 1;
    const int nxt = cur ^ 1;
    const bool more = (step + 1) < nsteps;
    if (more) {
      const int kN = (step + 1) << 5;
#pragma unroll
      for (int j = 0; j < 2; ++j) {
        const int row = ((tid * 2 + j) >> 2), seg = ((tid * 2 + j) & 3);
        async_cp16(Ab + (long long)row * lda + kN + seg * 8,
                   lA + nxt * LDS_A + row * BKP + seg * 8);
      }
#pragma unroll
      for (int j = 0; j < 4; ++j)
        async_cp16(Bb + (long long)bRow0 * ldb + kN + j * 8,
                   lB + nxt * LDS_B + bRow0 * BKP + j * 8);
    }

    const unsigned short* lAc = lA + cur * LDS_A;
    const unsigned short* lBc = lB + cur * LDS_B;
    FragBF aF[4], bF[4];
#pragma unroll
    for (int mi = 0; mi < 4; ++mi) {
      const uint4* p = (const uint4*)(lAc + (wm * 64 + mi * 16 + llo) * BKP);
      aF[mi].q[0] = p[lhi];
      aF[mi].q[1] = p[lhi + 2];
    }
#pragma unroll
    for (int ni = 0; ni < 4; ++ni) {
      const uint4* p = (const uint4*)(lBc + (wn * 64 + ni * 16 + llo) * BKP);
      bF[ni].q[0] = p[lhi];
      bF[ni].q[1] = p[lhi + 2];
    }
#pragma unroll
    for (int mi = 0; mi < 4; ++mi)
#pragma unroll
      for (int ni = 0; ni < 4; ++ni)
        acc[mi][ni] = __builtin_amdgcn_wmma_f32_16x16x32_bf16(
            false, aF[mi].v, false, bF[ni].v, (short)0, acc[mi][ni], false, false);

    if (more) wait_async0();
    __syncthreads();
  }
#else
  // ---- fallback: register-staged double-buffered pipeline ----
  uint4 ra[2], rb[4];
#pragma unroll
  for (int j = 0; j < 2; ++j) {
    const int row = ((tid * 2 + j) >> 2), seg = ((tid * 2 + j) & 3);
    ra[j] = *(const uint4*)(Ab + (long long)row * lda + seg * 8);
  }
#pragma unroll
  for (int j = 0; j < 4; ++j)
    rb[j] = *(const uint4*)(Bb + (long long)bRow0 * ldb + j * 8);
#pragma unroll
  for (int j = 0; j < 2; ++j) {
    const int row = ((tid * 2 + j) >> 2), seg = ((tid * 2 + j) & 3);
    *(uint4*)(lA + row * BKP + seg * 8) = ra[j];
  }
#pragma unroll
  for (int j = 0; j < 4; ++j)
    *(uint4*)(lB + bRow0 * BKP + j * 8) = rb[j];
  __syncthreads();

  for (int step = 0; step < nsteps; ++step) {
    const int cur = step & 1;
    const int nxt = cur ^ 1;
    const bool more = (step + 1) < nsteps;
    if (more) {
      const int kN = (step + 1) << 5;
#pragma unroll
      for (int j = 0; j < 2; ++j) {
        const int row = ((tid * 2 + j) >> 2), seg = ((tid * 2 + j) & 3);
        ra[j] = *(const uint4*)(Ab + (long long)row * lda + kN + seg * 8);
      }
#pragma unroll
      for (int j = 0; j < 4; ++j)
        rb[j] = *(const uint4*)(Bb + (long long)bRow0 * ldb + kN + j * 8);
    }

    const unsigned short* lAc = lA + cur * LDS_A;
    const unsigned short* lBc = lB + cur * LDS_B;
    FragBF aF[4], bF[4];
#pragma unroll
    for (int mi = 0; mi < 4; ++mi) {
      const uint4* p = (const uint4*)(lAc + (wm * 64 + mi * 16 + llo) * BKP);
      aF[mi].q[0] = p[lhi];
      aF[mi].q[1] = p[lhi + 2];
    }
#pragma unroll
    for (int ni = 0; ni < 4; ++ni) {
      const uint4* p = (const uint4*)(lBc + (wn * 64 + ni * 16 + llo) * BKP);
      bF[ni].q[0] = p[lhi];
      bF[ni].q[1] = p[lhi + 2];
    }
#pragma unroll
    for (int mi = 0; mi < 4; ++mi)
#pragma unroll
      for (int ni = 0; ni < 4; ++ni)
        acc[mi][ni] = __builtin_amdgcn_wmma_f32_16x16x32_bf16(
            false, aF[mi].v, false, bF[ni].v, (short)0, acc[mi][ni], false, false);

    if (more) {
#pragma unroll
      for (int j = 0; j < 2; ++j) {
        const int row = ((tid * 2 + j) >> 2), seg = ((tid * 2 + j) & 3);
        *(uint4*)(lA + nxt * LDS_A + row * BKP + seg * 8) = ra[j];
      }
#pragma unroll
      for (int j = 0; j < 4; ++j)
        *(uint4*)(lB + nxt * LDS_B + bRow0 * BKP + j * 8) = rb[j];
    }
    __syncthreads();
  }
#endif

  const int mBase = blockIdx.y * BM + wm * 64;
  const int nBase = blockIdx.x * BN + wn * 64;

#pragma unroll
  for (int mi = 0; mi < 4; ++mi) {
#pragma unroll
    for (int ni = 0; ni < 4; ++ni) {
      union { v8f v; float f[8]; } c;
      c.v = acc[mi][ni];
      const int n  = nBase + ni * 16 + llo;       // column
      const int m0 = mBase + mi * 16 + 8 * lhi;   // first of 8 rows
      if (EPI == 0) {
        unsigned short* O = (unsigned short*)OutP + bz * sC;
#pragma unroll
        for (int r = 0; r < 8; ++r)
          O[(long long)(m0 + r) * ldc + n] = f2bf(c.f[r]);
      } else if (EPI == 1) {
        unsigned short* O = (unsigned short*)OutP + bz * sC;
        uint4 pk;
        pk.x = (unsigned)f2bf(c.f[0]) | ((unsigned)f2bf(c.f[1]) << 16);
        pk.y = (unsigned)f2bf(c.f[2]) | ((unsigned)f2bf(c.f[3]) << 16);
        pk.z = (unsigned)f2bf(c.f[4]) | ((unsigned)f2bf(c.f[5]) << 16);
        pk.w = (unsigned)f2bf(c.f[6]) | ((unsigned)f2bf(c.f[7]) << 16);
        *(uint4*)(O + (long long)n * ldc + m0) = pk;   // 8 contiguous bf16
      } else {
        float* O       = (float*)OutP + bz * sC;
        const float* X = Xres + bz * sC;
        const float bias = bp[n];
#pragma unroll
        for (int r = 0; r < 8; ++r) {
          const long long idx = (long long)(m0 + r) * ldc + n;
          const float xv = __builtin_nontemporal_load(&X[idx]);
          __builtin_nontemporal_store(c.f[r] + bias + xv, &O[idx]);
        }
      }
    }
  }
}

// ---------------------------------------------------------------------------
extern "C" void kernel_launch(void* const* d_in, const int* in_sizes, int n_in,
                              void* d_out, int out_size, void* d_ws, size_t ws_size,
                              hipStream_t stream) {
  (void)in_sizes; (void)n_in; (void)out_size; (void)ws_size;
  constexpr int B = 8, S = 4096, D = 1024;

  const float* x  = (const float*)d_in[0];
  const float* Wq = (const float*)d_in[1];
  const float* Wk = (const float*)d_in[2];
  const float* Wv = (const float*)d_in[3];
  const float* Wp = (const float*)d_in[4];
  const float* bp = (const float*)d_in[5];

  const long long nBSD = (long long)B * S * D;
  const long long nDD  = (long long)D * D;

  char* ws = (char*)d_ws;
  size_t off = 0;
  unsigned short* xb  = (unsigned short*)(ws + off); off += (size_t)nBSD * 2;  // reused as z
  unsigned short* q   = (unsigned short*)(ws + off); off += (size_t)nBSD * 2;
  unsigned short* kT  = (unsigned short*)(ws + off); off += (size_t)nBSD * 2;  // [B, D, S]
  unsigned short* vT  = (unsigned short*)(ws + off); off += (size_t)nBSD * 2;  // [B, D, S]
  unsigned short* kvT = (unsigned short*)(ws + off); off += (size_t)B * D * D * 2;
  unsigned short* Wqb = (unsigned short*)(ws + off); off += (size_t)nDD * 2;
  unsigned short* Wkb = (unsigned short*)(ws + off); off += (size_t)nDD * 2;
  unsigned short* Wvb = (unsigned short*)(ws + off); off += (size_t)nDD * 2;
  unsigned short* Wpb = (unsigned short*)(ws + off); off += (size_t)nDD * 2;
  unsigned short* z   = xb;  // xb dead after QKV projections

  cvt_f32_to_bf16_x4<<<4096, 256, 0, stream>>>((const float4*)x,  (uint2*)xb,  nBSD / 4);
  cvt_f32_to_bf16_x4<<<512,  256, 0, stream>>>((const float4*)Wq, (uint2*)Wqb, nDD / 4);
  cvt_f32_to_bf16_x4<<<512,  256, 0, stream>>>((const float4*)Wk, (uint2*)Wkb, nDD / 4);
  cvt_f32_to_bf16_x4<<<512,  256, 0, stream>>>((const float4*)Wv, (uint2*)Wvb, nDD / 4);
  cvt_f32_to_bf16_x4<<<512,  256, 0, stream>>>((const float4*)Wp, (uint2*)Wpb, nDD / 4);

  const dim3 blk(256);
  const dim3 gSD(D / BN, S / BM, B);  // (4, 32, 8)
  const dim3 gDD(D / BN, D / BM, B);  // (4,  8, 8)
  const long long SD = (long long)S * D;
  const long long DS = (long long)D * S;
  const long long DD = (long long)D * D;

  // q = x @ Wq^T                       (bf16 row-major [B,S,D])
  gemm_bf16_wmma<0><<<gSD, blk, 0, stream>>>(xb, Wqb, q,   nullptr, nullptr,
                                             D, D, D, D, SD, 0, SD);
  // kT = (x @ Wk^T)^T                  (bf16 [B,D,S])
  gemm_bf16_wmma<1><<<gSD, blk, 0, stream>>>(xb, Wkb, kT,  nullptr, nullptr,
                                             D, D, D, S, SD, 0, DS);
  // vT = (x @ Wv^T)^T                  (bf16 [B,D,S])
  gemm_bf16_wmma<1><<<gSD, blk, 0, stream>>>(xb, Wvb, vT,  nullptr, nullptr,
                                             D, D, D, S, SD, 0, DS);
  // kvT[f,e] = sum_s vT[f,s] * kT[e,s]   ( = (K^T V)^T, bf16 [B,D,D])
  gemm_bf16_wmma<0><<<gDD, blk, 0, stream>>>(vT, kT, kvT,  nullptr, nullptr,
                                             S, S, S, D, DS, DS, DD);
  // z[s,f] = sum_e q[s,e] * kvT[f,e]     ( = Q (K^T V), bf16 [B,S,D])
  gemm_bf16_wmma<0><<<gSD, blk, 0, stream>>>(q, kvT, z,    nullptr, nullptr,
                                             D, D, D, D, SD, DD, SD);
  // out = z @ Wp^T + bp + x              (fp32 [B,S,D])
  gemm_bf16_wmma<2><<<gSD, blk, 0, stream>>>(z, Wpb, d_out, x, bp,
                                             D, D, D, D, SD, 0, SD);
}
